// MultiHeadAttention_4294967296304
// MI455X (gfx1250) — compile-verified, matrix operand order checked by reading, not run on hardware
//
#include <hip/hip_runtime.h>

// ---------------------------------------------------------------------------
// MHA block for MI455X (gfx1250, wave32, WMMA).
// B=4, S=2048, D=1024, H=8, DH=128.
//   q = X@Wq ; k = X@Wk ; v = X@Wv            (bf16 WMMA, f32 accum)
//   flash attention per (b,h), scale = 1/sqrt(D) = 1/32
//   o = q + attn ;  out = o + relu(o@Wo + bo)
// ---------------------------------------------------------------------------

typedef __attribute__((ext_vector_type(16))) __bf16        v16bf;
typedef __attribute__((ext_vector_type(8)))  float         v8f;
typedef __attribute__((ext_vector_type(4)))  float         v4f;
typedef __attribute__((ext_vector_type(4)))  unsigned int  v4u;
typedef __attribute__((ext_vector_type(2)))  unsigned int  v2u;

constexpr int Bb  = 4, Ss = 2048, Dd = 1024, Hh = 8, DHd = 128;
constexpr int Mm  = Bb * Ss;          // 8192 rows
constexpr float kScale = 0.03125f;    // 1/sqrt(1024)

union FragU { v4u u[2]; v16bf v; };

__device__ __forceinline__ unsigned short f2bf(float f) {
    unsigned int u = __float_as_uint(f);
    u += 0x7FFFu + ((u >> 16) & 1u);   // round-to-nearest-even
    return (unsigned short)(u >> 16);
}

// 16 contiguous bf16 (32 bytes, 16B-aligned) -> WMMA fragment
__device__ __forceinline__ v16bf ld_frag(const unsigned short* p) {
    FragU fu;
    fu.u[0] = *(const v4u*)(p);
    fu.u[1] = *(const v4u*)(p + 8);
    return fu.v;
}

__device__ __forceinline__ v8f zero_v8() {
    v8f z;
#pragma unroll
    for (int r = 0; r < 8; ++r) z[r] = 0.0f;
    return z;
}

// ---------------------------------------------------------------------------
// GEMM: C[M=8192, N=1024] = A[f32] * W[f32], bf16 WMMA with f32 accumulate.
// Block tile 128x128, 8 waves of 32x64 (2x4 16x16 frags), K-step 32.
// Per k-step: 8 WMMA vs 6 fragment loads.
// MODE 0: projection  -> bf16 out in [B][H][S][DH] layout (+ optional f32 copy)
// MODE 1: out-proj    -> outF = resid + relu(acc + bias)
// ---------------------------------------------------------------------------
constexpr int BM = 128, BN = 128, BK = 32;
constexpr int LDA_S = 40;   // padded LDS stride (elems): rows 80B -> 16B aligned frags

template <int MODE>
__global__ __launch_bounds__(256)
void gemm_bf16(const float* __restrict__ A, const float* __restrict__ W,
               const float* __restrict__ bias, const float* __restrict__ resid,
               unsigned short* __restrict__ outBf, float* __restrict__ outF) {
    __shared__ unsigned short As[BM * LDA_S];   // 128 x 32 (k-contiguous)
    __shared__ unsigned short Bs[BN * LDA_S];   // 128(n) x 32(k), N-major

    const int tid  = threadIdx.x;
    const int lane = tid & 31, wid = tid >> 5;
    const int hl   = lane >> 4, lr = lane & 15;      // half, lane-in-half
    const int m0 = blockIdx.y * BM, n0 = blockIdx.x * BN;
    const int waveM = (wid >> 1) * 32, waveN = (wid & 1) * 64;

    v8f acc[2][4];
#pragma unroll
    for (int i = 0; i < 2; ++i)
#pragma unroll
        for (int j = 0; j < 4; ++j) acc[i][j] = zero_v8();

    for (int k0 = 0; k0 < Dd; k0 += BK) {
        __syncthreads();
        // Stage A tile 128x32: f32 -> bf16
#pragma unroll
        for (int i = 0; i < 4; ++i) {
            int g = tid + i * 256;
            int row = g >> 3, col = (g & 7) * 4;
            const float* ap = A + (size_t)(m0 + row) * Dd + k0 + col;
            v4f f = *(const v4f*)ap;
            if (k0 + BK < Dd) __builtin_prefetch(ap + BK, 0, 1);
            v2u pk;
            pk[0] = (unsigned int)f2bf(f[0]) | ((unsigned int)f2bf(f[1]) << 16);
            pk[1] = (unsigned int)f2bf(f[2]) | ((unsigned int)f2bf(f[3]) << 16);
            *(v2u*)(&As[row * LDA_S + col]) = pk;
        }
        // Stage W tile 32(k)x128(n), transposed into Bs[n][k]
#pragma unroll
        for (int i = 0; i < 4; ++i) {
            int g = tid + i * 256;
            int kr = g >> 5, nc = (g & 31) * 4;
            const float* wp = W + (size_t)(k0 + kr) * Dd + n0 + nc;
            v4f f = *(const v4f*)wp;
            if (k0 + BK < Dd) __builtin_prefetch(wp + (size_t)BK * Dd, 0, 1);
#pragma unroll
            for (int j = 0; j < 4; ++j)
                Bs[(nc + j) * LDA_S + kr] = f2bf(f[j]);
        }
        __syncthreads();

        v16bf af[2], bfr[4];
#pragma unroll
        for (int i = 0; i < 2; ++i)
            af[i] = ld_frag(&As[(waveM + i * 16 + lr) * LDA_S + hl * 16]);
#pragma unroll
        for (int j = 0; j < 4; ++j)
            bfr[j] = ld_frag(&Bs[(waveN + j * 16 + lr) * LDA_S + hl * 16]);
#pragma unroll
        for (int i = 0; i < 2; ++i)
#pragma unroll
            for (int j = 0; j < 4; ++j)
                acc[i][j] = __builtin_amdgcn_wmma_f32_16x16x32_bf16(
                    false, af[i], false, bfr[j], (short)0, acc[i][j], false, false);
    }

    // Epilogue (C layout: VGPR r -> M = r + 8*half, N = lane%16)
#pragma unroll
    for (int i = 0; i < 2; ++i) {
#pragma unroll
        for (int j = 0; j < 4; ++j) {
#pragma unroll
            for (int r = 0; r < 8; ++r) {
                int m = m0 + waveM + i * 16 + r + hl * 8;
                int n = n0 + waveN + j * 16 + lr;
                float v = acc[i][j][r];
                if constexpr (MODE == 0) {
                    int bb = m / Ss, s = m % Ss;
                    int h  = n / DHd, d = n % DHd;
                    outBf[(((size_t)(bb * Hh + h)) * Ss + s) * DHd + d] = f2bf(v);
                    if (outF) outF[(size_t)m * Dd + n] = v;
                } else {
                    float t = v + bias[n];
                    outF[(size_t)m * Dd + n] =
                        resid[(size_t)m * Dd + n] + fmaxf(t, 0.0f);
                }
            }
        }
    }
}

// ---------------------------------------------------------------------------
// Flash attention per (b,h): 128 query rows / block (16 per wave),
// key blocks of 32, online softmax, all matmuls on bf16 WMMA.
// K tile staged with CDNA5 async global->LDS DMA (ASYNCcnt).
// ---------------------------------------------------------------------------
constexpr int KS_S = 136;   // K tile LDS stride (dh elems), padded
constexpr int VT_S = 40;    // V-transposed stride (key elems), padded
constexpr int P_S  = 40;    // P tile stride, padded

__global__ __launch_bounds__(256)
void attention_kernel(const unsigned short* __restrict__ Qh,
                      const unsigned short* __restrict__ Kh,
                      const unsigned short* __restrict__ Vh,
                      const float* __restrict__ qres,
                      float* __restrict__ o) {
    __shared__ unsigned short Ks[32 * KS_S];          // 32 keys x 128 dh
    __shared__ unsigned short Vt[DHd * VT_S];         // 128 dh  x 32 keys
    __shared__ unsigned short Ps[8 * 16 * P_S];       // per-wave 16x32 P

    const int tid  = threadIdx.x;
    const int lane = tid & 31, wid = tid >> 5;
    const int hl   = lane >> 4, lr = lane & 15;
    const int bh = blockIdx.y;
    const int b = bh / Hh, h = bh % Hh;
    const size_t headOff = ((size_t)(b * Hh + h)) * Ss * DHd;
    const unsigned short* Kbase = Kh + headOff;
    const unsigned short* Vbase = Vh + headOff;
    const int q0 = blockIdx.x * 128;

    // Preload Q fragments (16 rows x 128 dh = 4 frags), held in VGPRs
    const unsigned short* qptr = Qh + headOff + (size_t)(q0 + wid * 16 + lr) * DHd;
    v16bf qf[4];
#pragma unroll
    for (int i = 0; i < 4; ++i)
        qf[i] = ld_frag(qptr + i * 32 + hl * 16);

    float mstate[8], lstate[8];
    v8f Oacc[8];
#pragma unroll
    for (int r = 0; r < 8; ++r) { mstate[r] = -3.0e38f; lstate[r] = 0.0f; }
#pragma unroll
    for (int c = 0; c < 8; ++c) Oacc[c] = zero_v8();

    unsigned short* Pw = &Ps[wid * 16 * P_S];

    for (int kb = 0; kb < Ss; kb += 32) {
        __syncthreads();
        // Stage K block via async global->LDS DMA (16B per lane, no VGPR hop)
#pragma unroll
        for (int i = 0; i < 2; ++i) {
            int g = tid + i * 256;
            int key = g >> 4, d8 = (g & 15) * 8;
            unsigned lds_off = (unsigned)(size_t)(&Ks[key * KS_S + d8]);
            const unsigned short* gp = Kbase + (size_t)(kb + key) * DHd + d8;
            asm volatile("global_load_async_to_lds_b128 %0, %1, off"
                         :: "v"(lds_off),
                            "v"((unsigned long long)(size_t)gp)
                         : "memory");
            if (kb + 32 < Ss) __builtin_prefetch(gp + 32 * DHd, 0, 1);
        }
        // Stage V block transposed: Vt[dh][key]
        {
            int key = tid & 31, d0 = (tid >> 5) * 16;
            const unsigned short* vp = Vbase + (size_t)(kb + key) * DHd + d0;
            union { v4u u[2]; unsigned short s[16]; } vv;
            vv.u[0] = *(const v4u*)vp;
            vv.u[1] = *(const v4u*)(vp + 8);
            if (kb + 32 < Ss) __builtin_prefetch(vp + 32 * DHd, 0, 1);
#pragma unroll
            for (int t = 0; t < 16; ++t)
                Vt[(d0 + t) * VT_S + key] = vv.s[t];
        }
        asm volatile("s_wait_asynccnt 0x0" ::: "memory");
        __syncthreads();

        // Logits: two 16x16 tiles (keys kb..+15, kb+16..+31), contraction DH=128
        v8f l0 = zero_v8(), l1 = zero_v8();
#pragma unroll
        for (int d = 0; d < 4; ++d) {
            v16bf kf0 = ld_frag(&Ks[lr * KS_S        + d * 32 + hl * 16]);
            v16bf kf1 = ld_frag(&Ks[(16 + lr) * KS_S + d * 32 + hl * 16]);
            l0 = __builtin_amdgcn_wmma_f32_16x16x32_bf16(false, qf[d], false, kf0,
                                                         (short)0, l0, false, false);
            l1 = __builtin_amdgcn_wmma_f32_16x16x32_bf16(false, qf[d], false, kf1,
                                                         (short)0, l1, false, false);
        }

        // Online softmax (row m = r + 8*half lives across 16 lanes of a half)
        float corrv[8];
#pragma unroll
        for (int r = 0; r < 8; ++r) {
            float x0 = l0[r] * kScale, x1 = l1[r] * kScale;
            float t = fmaxf(x0, x1);
            t = fmaxf(t, __shfl_xor(t, 1, 32));
            t = fmaxf(t, __shfl_xor(t, 2, 32));
            t = fmaxf(t, __shfl_xor(t, 4, 32));
            t = fmaxf(t, __shfl_xor(t, 8, 32));
            float mnew = fmaxf(mstate[r], t);
            float p0 = __expf(x0 - mnew), p1 = __expf(x1 - mnew);
            float rs = p0 + p1;
            rs += __shfl_xor(rs, 1, 32);
            rs += __shfl_xor(rs, 2, 32);
            rs += __shfl_xor(rs, 4, 32);
            rs += __shfl_xor(rs, 8, 32);
            float corr = __expf(mstate[r] - mnew);
            corrv[r] = corr;
            lstate[r] = lstate[r] * corr + rs;
            mstate[r] = mnew;
            int mrow = r + hl * 8;                 // C-layout row
            Pw[mrow * P_S + lr]      = f2bf(p0);   // tile0 cols 0..15
            Pw[mrow * P_S + 16 + lr] = f2bf(p1);   // tile1 cols 16..31
        }
        // Per-wave LDS ops are in-order; fence compiler + DS pipeline
        asm volatile("s_wait_dscnt 0x0" ::: "memory");

        // O += P @ V  (A-frag of P from LDS, B-frag of V from transposed LDS)
        v16bf pf = ld_frag(&Pw[lr * P_S + hl * 16]);
#pragma unroll
        for (int c = 0; c < 8; ++c) {
#pragma unroll
            for (int r = 0; r < 8; ++r) Oacc[c][r] *= corrv[r];
            v16bf vf = ld_frag(&Vt[(c * 16 + lr) * VT_S + hl * 16]);
            Oacc[c] = __builtin_amdgcn_wmma_f32_16x16x32_bf16(
                false, pf, false, vf, (short)0, Oacc[c], false, false);
        }
    }

    // Normalize, add projected-query residual, write merged-head f32
    float invl[8];
#pragma unroll
    for (int r = 0; r < 8; ++r) invl[r] = 1.0f / lstate[r];
#pragma unroll
    for (int c = 0; c < 8; ++c) {
#pragma unroll
        for (int r = 0; r < 8; ++r) {
            int m   = b * Ss + q0 + wid * 16 + r + hl * 8;
            int col = h * DHd + c * 16 + lr;
            size_t idx = (size_t)m * Dd + col;
            o[idx] = qres[idx] + Oacc[c][r] * invl[r];
        }
    }
}

// ---------------------------------------------------------------------------
extern "C" void kernel_launch(void* const* d_in, const int* in_sizes, int n_in,
                              void* d_out, int out_size, void* d_ws, size_t ws_size,
                              hipStream_t stream) {
    const float* query = (const float*)d_in[0];
    const float* key   = (const float*)d_in[1];
    const float* value = (const float*)d_in[2];
    const float* Wq    = (const float*)d_in[3];
    const float* Wk    = (const float*)d_in[4];
    const float* Wv    = (const float*)d_in[5];
    const float* Wo    = (const float*)d_in[6];
    const float* bo    = (const float*)d_in[7];

    // Workspace carve-out (total 112 MB):
    char* ws = (char*)d_ws;
    float*          q_f32 = (float*)(ws);                         // 33.5 MB
    float*          o_f32 = (float*)(ws + 33554432);              // 33.5 MB
    unsigned short* Qh    = (unsigned short*)(ws + 67108864);     // 16.8 MB bf16
    unsigned short* Kh    = (unsigned short*)(ws + 83886080);     // 16.8 MB
    unsigned short* Vh    = (unsigned short*)(ws + 100663296);    // 16.8 MB

    dim3 ggrid(Dd / BN, Mm / BM);   // (8, 64)
    gemm_bf16<0><<<ggrid, 256, 0, stream>>>(query, Wq, nullptr, nullptr, Qh, q_f32);
    gemm_bf16<0><<<ggrid, 256, 0, stream>>>(key,   Wk, nullptr, nullptr, Kh, nullptr);
    gemm_bf16<0><<<ggrid, 256, 0, stream>>>(value, Wv, nullptr, nullptr, Vh, nullptr);

    dim3 agrid(Ss / 128, Bb * Hh);  // (16, 32)
    attention_kernel<<<agrid, 256, 0, stream>>>(Qh, Kh, Vh, q_f32, o_f32);

    gemm_bf16<1><<<ggrid, 256, 0, stream>>>(o_f32, Wo, bo, o_f32,
                                            (unsigned short*)nullptr, (float*)d_out);
}
